// TreeDecoder_40200893890812
// MI455X (gfx1250) — matrix operand order repaired and checked by reading, not compiled, not run on hardware
//
#include <hip/hip_runtime.h>
#include <hip/hip_bf16.h>

// ---------------- problem constants ----------------
#define BB 32
#define TT 32
#define LL 64
#define DD 1024
#define WV 32000
#define VV 32500

typedef __attribute__((ext_vector_type(16))) __bf16 v16bf;
typedef __attribute__((ext_vector_type(8)))  float  v8f;

struct Frag16 { union { uint4 q[2]; v16bf v; }; };

__device__ __forceinline__ unsigned short f2bf(float x) {
    unsigned u = __float_as_uint(x);
    unsigned r = (u + 0x7FFFu + ((u >> 16) & 1u)) >> 16;
    return (unsigned short)r;
}
__device__ __forceinline__ float neg_inf() { return __int_as_float(0xff800000); }
__device__ __forceinline__ float sigm(float x) { return 1.f / (1.f + __expf(-x)); }

__device__ __forceinline__ v8f wmma_bf16(v16bf a, v16bf b, v8f c) {
    return __builtin_amdgcn_wmma_f32_16x16x32_bf16(false, a, false, b, (short)0, c, false, false);
}

// CDNA5 async memory->LDS (ISA 15.18.3 opc 96-98, ASYNCcnt). LDS address is the
// low 32 bits of the generic pointer (LDS aperture lives in ADDR[63:32]).
__device__ __forceinline__ void async_ld_b64(unsigned lds_addr, const void* gaddr) {
    asm volatile("global_load_async_to_lds_b64 %0, %1, off"
                 :: "v"(lds_addr), "v"(gaddr) : "memory");
}
__device__ __forceinline__ void async_ld_b128(unsigned lds_addr, const void* gaddr) {
    asm volatile("global_load_async_to_lds_b128 %0, %1, off"
                 :: "v"(lds_addr), "v"(gaddr) : "memory");
}

// ---------------- conversion / setup kernels ----------------
__global__ void k_cvt_bf16(const float* __restrict__ src, unsigned short* __restrict__ dst, size_t n) {
    size_t i = (size_t)blockIdx.x * blockDim.x + threadIdx.x;
    size_t stride = (size_t)gridDim.x * blockDim.x;
    for (; i < n; i += stride) dst[i] = f2bf(src[i]);
}

__global__ void k_gather_emb(const int* __restrict__ y_train, const float* __restrict__ emb,
                             unsigned short* __restrict__ yemb) {
    size_t i = (size_t)blockIdx.x * 256 + threadIdx.x; // (t*B + b)*D + d
    int d = (int)(i & (DD - 1));
    size_t tb = i >> 10;
    int b = (int)(tb & (BB - 1));
    int t = (int)(tb >> 5);
    int tok = y_train[(b * TT + t) * 3 + 0];
    yemb[i] = f2bf(emb[(size_t)tok * DD + d]);
}

__global__ void k_init(const float* __restrict__ h0, const float* __restrict__ c0,
                       float* __restrict__ rh, float* __restrict__ rc,
                       float* __restrict__ wh, float* __restrict__ wc,
                       unsigned short* __restrict__ rh_bf, unsigned short* __restrict__ wh_bf,
                       float* __restrict__ rfeed, float* __restrict__ wfeed) {
    int i = blockIdx.x * 256 + threadIdx.x;
    if (i < BB * DD) {
        float h = h0[i], c = c0[i];
        rh[i] = h; wh[i] = h; rc[i] = c; wc[i] = c;
        unsigned short hb = f2bf(h);
        rh_bf[i] = hb; wh_bf[i] = hb;
    }
    if (i < BB * 2 * DD) { rfeed[i] = 0.f; wfeed[i] = 0.f; }
}

// ---------------- concat kernels (build bf16 GEMM inputs) ----------------
__global__ void k_concat_word(const unsigned short* __restrict__ yemb_t,
                              const unsigned short* __restrict__ rh_bf,
                              const float* __restrict__ wfeed,
                              unsigned short* __restrict__ xw) {
    int i = blockIdx.x * 256 + threadIdx.x;           // B*4096
    int b = i >> 12, col = i & 4095;
    unsigned short v;
    if (col < DD)            v = yemb_t[b * DD + col];
    else if (col < 2 * DD)   v = rh_bf[b * DD + (col - DD)];
    else                     v = f2bf(wfeed[b * 2 * DD + (col - 2 * DD)]);
    xw[i] = v;
}

__global__ void k_concat_rule(const unsigned short* __restrict__ yemb_t,
                              const float* __restrict__ rfeed,
                              const unsigned short* __restrict__ wh_bf,
                              unsigned short* __restrict__ xr) {
    int i = blockIdx.x * 256 + threadIdx.x;           // B*4096
    int b = i >> 12, col = i & 4095;
    unsigned short v;
    if (col < DD)            v = yemb_t[b * DD + col];
    else if (col < 3 * DD)   v = f2bf(rfeed[b * 2 * DD + (col - DD)]);
    else                     v = wh_bf[b * DD + (col - 3 * DD)];
    xr[i] = v;
}

__global__ void k_concat_inp(const unsigned short* __restrict__ rh_bf,
                             const float* __restrict__ rctx,
                             const unsigned short* __restrict__ wh_bf,
                             const float* __restrict__ wctx,
                             unsigned short* __restrict__ inp) {
    int i = blockIdx.x * 256 + threadIdx.x;           // B*6144
    int b = i / 6144, col = i % 6144;
    unsigned short v;
    if (col < DD)            v = rh_bf[b * DD + col];
    else if (col < 3 * DD)   v = f2bf(rctx[b * 2 * DD + (col - DD)]);
    else if (col < 4 * DD)   v = wh_bf[b * DD + (col - 3 * DD)];
    else                     v = f2bf(wctx[b * 2 * DD + (col - 4 * DD)]);
    inp[i] = v;
}

// ---------------- skinny WMMA GEMM:  C[32,N] (+)= A[32,K] * W[N,K]^T ----------------
// One wave per 16-col tile; each wave computes both 16-row M tiles so each bf16
// weight row is streamed exactly once (L2-resident after step 0; 169MB < 192MB L2).
__global__ void k_gemm_skinny(const unsigned short* __restrict__ A,
                              const unsigned short* __restrict__ W,
                              float* __restrict__ C, int K, int N, int accum) {
    int wave = threadIdx.x >> 5;
    int lane = threadIdx.x & 31;
    int n0 = (blockIdx.x * 4 + wave) * 16;
    if (n0 >= N) return;
    int g = lane >> 4, r = lane & 15;
    v8f acc0 = {}, acc1 = {};
    const unsigned short* wrow  = W + (size_t)(n0 + r) * K;
    const unsigned short* arow0 = A + (size_t)r * K;
    const unsigned short* arow1 = A + (size_t)(r + 16) * K;
    for (int k0 = 0; k0 < K; k0 += 32) {
        __builtin_prefetch(wrow + k0 + 512, 0, 0);    // global_prefetch_b8 (speculative)
        Frag16 bf, a0, a1;
        bf.q[0] = *(const uint4*)(wrow + k0 + 8 * g);
        bf.q[1] = *(const uint4*)(wrow + k0 + 16 + 8 * g);
        a0.q[0] = *(const uint4*)(arow0 + k0 + 8 * g);
        a0.q[1] = *(const uint4*)(arow0 + k0 + 16 + 8 * g);
        a1.q[0] = *(const uint4*)(arow1 + k0 + 8 * g);
        a1.q[1] = *(const uint4*)(arow1 + k0 + 16 + 8 * g);
        acc0 = wmma_bf16(a0.v, bf.v, acc0);
        acc1 = wmma_bf16(a1.v, bf.v, acc1);
    }
    int col = n0 + r;
    #pragma unroll
    for (int v = 0; v < 8; v++) {
        int m = v + 8 * g;
        size_t i0 = (size_t)m * N + col;
        size_t i1 = (size_t)(m + 16) * N + col;
        if (accum) { C[i0] += acc0[v]; C[i1] += acc1[v]; }
        else       { C[i0]  = acc0[v]; C[i1]  = acc1[v]; }
    }
}

// ---------------- LSTM pointwise (gates -> h2,c2 with optional word-mask blend) ----------------
__global__ void k_lstm(const float* __restrict__ gsum, const float* __restrict__ bih,
                       const float* __restrict__ bhh,
                       float* __restrict__ h, float* __restrict__ c,
                       unsigned short* __restrict__ h_bf,
                       const int* __restrict__ y_train, int t, int use_mask) {
    int idx = blockIdx.x * 256 + threadIdx.x;         // B*D
    int b = idx >> 10, d = idx & (DD - 1);
    const float* gb = gsum + (size_t)b * 4 * DD;
    float gi = gb[d]            + bih[d]            + bhh[d];
    float gf = gb[DD + d]       + bih[DD + d]       + bhh[DD + d];
    float gg = gb[2 * DD + d]   + bih[2 * DD + d]   + bhh[2 * DD + d];
    float go = gb[3 * DD + d]   + bih[3 * DD + d]   + bhh[3 * DD + d];
    float ho = h[idx], co = c[idx];
    float c2 = sigm(gf) * co + sigm(gi) * tanhf(gg);
    float h2 = sigm(go) * tanhf(c2);
    if (use_mask) {
        float wm = (float)y_train[(b * TT + t) * 3 + 2];
        h2 = h2 * wm + ho * (1.f - wm);
        c2 = c2 * wm + co * (1.f - wm);
    }
    h[idx] = h2; c[idx] = c2; h_bf[idx] = f2bf(h2);
}

// ---------------- attention (L=64, tiny; one block per batch element) ----------------
__global__ void k_attention(const float* __restrict__ q, const float* __restrict__ keys,
                            const float* __restrict__ vals, const unsigned char* __restrict__ xmask,
                            float* __restrict__ ctx) {
    int b = blockIdx.x, tid = threadIdx.x;            // 256 threads
    __shared__ float part[256];
    __shared__ float w[LL];
    __shared__ float invsum;
    int l = tid >> 2, j = tid & 3;
    const float* kr = keys + ((size_t)b * LL + l) * DD;
    const float* qr = q + (size_t)b * DD;
    float s = 0.f;
    for (int d = j * 256; d < j * 256 + 256; d++) s += kr[d] * qr[d];
    part[tid] = s;
    __syncthreads();
    if (j == 0) {
        float sc = (part[tid] + part[tid + 1] + part[tid + 2] + part[tid + 3]) * 0.03125f;
        if (xmask[b * LL + l]) sc = neg_inf();
        w[l] = sc;
    }
    __syncthreads();
    if (tid == 0) {
        float mx = neg_inf();
        for (int i = 0; i < LL; i++) mx = fmaxf(mx, w[i]);
        float sum = 0.f;
        for (int i = 0; i < LL; i++) { float e = __expf(w[i] - mx); w[i] = e; sum += e; }
        invsum = 1.f / sum;
    }
    __syncthreads();
    float inv = invsum;
    for (int e = tid; e < 2 * DD; e += 256) {
        float acc = 0.f;
        for (int i = 0; i < LL; i++) acc += w[i] * vals[((size_t)b * LL + i) * 2 * DD + e];
        ctx[(size_t)b * 2 * DD + e] = acc * inv;
    }
}

__global__ void k_tanh_bf(const float* __restrict__ src, unsigned short* __restrict__ dst, int n) {
    int i = blockIdx.x * 256 + threadIdx.x;
    if (i < n) dst[i] = f2bf(tanhf(src[i]));
}

// ---------------- logits GEMM: out[t,b,:] = mask( P[1024,1024] * W[N,1024]^T ) ----------------
// Double-buffered async (memory->LDS, ASYNCcnt) WMMA pipeline.
// Block = 8 waves (2 M x 4 N), block tile 32x64, K chunk 32.
// Each thread issues exactly 2 async ops per chunk (A: b64, B: b128), so each
// wave carries a uniform 2 outstanding ASYNCcnt per staged chunk; in-order
// completion makes `s_wait_asynccnt 0x2` release the previous chunk.
__global__ void k_gemm_logits(const unsigned short* __restrict__ P,
                              const unsigned short* __restrict__ W,
                              float* __restrict__ out,
                              const int* __restrict__ score_mask,
                              int Nrows, int colOffset, int maskIfSmZero) {
    __shared__ unsigned short As[2][32 * 32];
    __shared__ unsigned short Bs[2][64 * 32];
    int tid = threadIdx.x;
    int wave = tid >> 5, lane = tid & 31;
    int wm = wave >> 2, wn = wave & 3;
    int g = lane >> 4, r = lane & 15;
    int m0 = blockIdx.y * 32;
    int n0 = blockIdx.x * 64;

    // staging coordinates: A 32x32 -> 4 halfs/thread, B 64x32 -> 8 halfs/thread
    int ea = tid * 4;
    int ra = ea >> 5, ca = ea & 31;
    int eb = tid * 8;
    int rb = eb >> 5, cb = eb & 31;
    const unsigned short* gA = P + (size_t)(m0 + ra) * DD + ca;
    // clamp OOB B rows (duplicate data; their columns are discarded in epilogue)
    int rbc = (n0 + rb < Nrows) ? (n0 + rb) : (Nrows - 1);
    const unsigned short* gB = W + (size_t)rbc * DD + cb;
    unsigned ldsA[2], ldsB[2];
    ldsA[0] = (unsigned)(size_t)&As[0][ea];
    ldsA[1] = (unsigned)(size_t)&As[1][ea];
    ldsB[0] = (unsigned)(size_t)&Bs[0][eb];
    ldsB[1] = (unsigned)(size_t)&Bs[1][eb];

    v8f acc = {};
    // prologue: stage chunk 0 into buffer 0
    async_ld_b64(ldsA[0], gA);
    async_ld_b128(ldsB[0], gB);
    for (int k0 = 0; k0 < DD; k0 += 32) {
        int buf = (k0 >> 5) & 1;
        if (k0 + 32 < DD) {
            async_ld_b64(ldsA[buf ^ 1], gA + k0 + 32);
            async_ld_b128(ldsB[buf ^ 1], gB + k0 + 32);
            asm volatile("s_wait_asynccnt 0x2" ::: "memory");  // chunk k0 complete
        } else {
            asm volatile("s_wait_asynccnt 0x0" ::: "memory");
        }
        __syncthreads();                                       // cross-wave LDS visibility
        Frag16 a, bf;
        const unsigned short* ar = &As[buf][(wm * 16 + r) * 32];
        a.q[0] = *(const uint4*)(ar + 8 * g);
        a.q[1] = *(const uint4*)(ar + 16 + 8 * g);
        const unsigned short* br = &Bs[buf][(wn * 16 + r) * 32];
        bf.q[0] = *(const uint4*)(br + 8 * g);
        bf.q[1] = *(const uint4*)(br + 16 + 8 * g);
        acc = wmma_bf16(a.v, bf.v, acc);
        __syncthreads();   // protect `buf` before it is re-staged next iteration
    }
    int col = n0 + wn * 16 + r;
    if (col < Nrows) {
        #pragma unroll
        for (int v = 0; v < 8; v++) {
            int mrow = m0 + wm * 16 + v + 8 * g;     // row = t*B + b
            int t = mrow >> 5, b = mrow & 31;
            int sm = score_mask[b * TT + t];
            bool masked = maskIfSmZero ? (sm == 0) : (sm != 0);
            out[((size_t)t * BB + b) * VV + colOffset + col] = masked ? neg_inf() : acc[v];
        }
    }
}

// ---------------- host launch ----------------
extern "C" void kernel_launch(void* const* d_in, const int* in_sizes, int n_in,
                              void* d_out, int out_size, void* d_ws, size_t ws_size,
                              hipStream_t stream) {
    const float* x_enc     = (const float*)d_in[0];
    const float* x_enc_k   = (const float*)d_in[1];
    const float* dec_h0    = (const float*)d_in[2];
    const float* dec_c0    = (const float*)d_in[3];
    const unsigned char* x_mask = (const unsigned char*)d_in[4];
    const int*   y_train   = (const int*)d_in[5];
    const int*   score_mask= (const int*)d_in[7];
    const float* emb_W     = (const float*)d_in[8];
    const float* rule_Wih  = (const float*)d_in[9];
    const float* rule_Whh  = (const float*)d_in[10];
    const float* rule_bih  = (const float*)d_in[11];
    const float* rule_bhh  = (const float*)d_in[12];
    const float* word_Wih  = (const float*)d_in[13];
    const float* word_Whh  = (const float*)d_in[14];
    const float* word_bih  = (const float*)d_in[15];
    const float* word_bhh  = (const float*)d_in[16];
    const float* rule_ro_W = (const float*)d_in[17];
    const float* word_ro_W = (const float*)d_in[18];
    const float* readout_W = (const float*)d_in[19];
    float* out = (float*)d_out;

    char* ws = (char*)d_ws;
    size_t off = 0;
    auto alloc = [&](size_t bytes) { size_t o = off; off = (off + bytes + 255) & ~(size_t)255; return o; };
    unsigned short* wWih_bf  = (unsigned short*)(ws + alloc((size_t)4 * DD * 4 * DD * 2));
    unsigned short* rWih_bf  = (unsigned short*)(ws + alloc((size_t)4 * DD * 4 * DD * 2));
    unsigned short* wWhh_bf  = (unsigned short*)(ws + alloc((size_t)4 * DD * DD * 2));
    unsigned short* rWhh_bf  = (unsigned short*)(ws + alloc((size_t)4 * DD * DD * 2));
    unsigned short* wro_bf   = (unsigned short*)(ws + alloc((size_t)DD * 6 * DD * 2));
    unsigned short* rro_bf   = (unsigned short*)(ws + alloc((size_t)DD * 6 * DD * 2));
    unsigned short* rout_bf  = (unsigned short*)(ws + alloc((size_t)VV * DD * 2));
    unsigned short* yemb_bf  = (unsigned short*)(ws + alloc((size_t)TT * BB * DD * 2));
    unsigned short* xw_bf    = (unsigned short*)(ws + alloc((size_t)BB * 4 * DD * 2));
    unsigned short* xr_bf    = (unsigned short*)(ws + alloc((size_t)BB * 4 * DD * 2));
    unsigned short* inp_bf   = (unsigned short*)(ws + alloc((size_t)BB * 6 * DD * 2));
    float* gbuf  = (float*)(ws + alloc((size_t)BB * 4 * DD * 4));
    float* gro   = (float*)(ws + alloc((size_t)BB * DD * 4));
    float* rh    = (float*)(ws + alloc((size_t)BB * DD * 4));
    float* rc    = (float*)(ws + alloc((size_t)BB * DD * 4));
    float* wh    = (float*)(ws + alloc((size_t)BB * DD * 4));
    float* wc    = (float*)(ws + alloc((size_t)BB * DD * 4));
    unsigned short* rh_bf = (unsigned short*)(ws + alloc((size_t)BB * DD * 2));
    unsigned short* wh_bf = (unsigned short*)(ws + alloc((size_t)BB * DD * 2));
    float* rfeed = (float*)(ws + alloc((size_t)BB * 2 * DD * 4));
    float* wfeed = (float*)(ws + alloc((size_t)BB * 2 * DD * 4));
    unsigned short* rpre_bf = (unsigned short*)(ws + alloc((size_t)TT * BB * DD * 2));
    unsigned short* wpre_bf = (unsigned short*)(ws + alloc((size_t)TT * BB * DD * 2));
    (void)ws_size; (void)n_in; (void)in_sizes; (void)out_size;

    // weights -> bf16 (L2-resident working set afterwards: ~169MB < 192MB L2)
    k_cvt_bf16<<<4096, 256, 0, stream>>>(word_Wih, wWih_bf, (size_t)4 * DD * 4 * DD);
    k_cvt_bf16<<<4096, 256, 0, stream>>>(rule_Wih, rWih_bf, (size_t)4 * DD * 4 * DD);
    k_cvt_bf16<<<2048, 256, 0, stream>>>(word_Whh, wWhh_bf, (size_t)4 * DD * DD);
    k_cvt_bf16<<<2048, 256, 0, stream>>>(rule_Whh, rWhh_bf, (size_t)4 * DD * DD);
    k_cvt_bf16<<<2048, 256, 0, stream>>>(word_ro_W, wro_bf, (size_t)DD * 6 * DD);
    k_cvt_bf16<<<2048, 256, 0, stream>>>(rule_ro_W, rro_bf, (size_t)DD * 6 * DD);
    k_cvt_bf16<<<8192, 256, 0, stream>>>(readout_W, rout_bf, (size_t)VV * DD);
    k_gather_emb<<<(TT * BB * DD) / 256, 256, 0, stream>>>(y_train, emb_W, yemb_bf);
    k_init<<<(BB * 2 * DD) / 256, 256, 0, stream>>>(dec_h0, dec_c0, rh, rc, wh, wc,
                                                    rh_bf, wh_bf, rfeed, wfeed);

    const int gElem    = (BB * 4 * DD) / 256;   // concat grids
    const int gInp     = (BB * 6 * DD) / 256;
    const int gLstm    = (BB * DD) / 256;
    const int gSkinny4 = (4 * DD / 16) / 4;     // N=4096 -> 64 blocks of 4 waves
    const int gSkinny1 = (DD / 16) / 4;         // N=1024 -> 16 blocks

    for (int t = 0; t < TT; t++) {
        const unsigned short* yemb_t = yemb_bf + (size_t)t * BB * DD;
        // word LSTM
        k_concat_word<<<gElem, 256, 0, stream>>>(yemb_t, rh_bf, wfeed, xw_bf);
        k_gemm_skinny<<<gSkinny4, 128, 0, stream>>>(xw_bf, wWih_bf, gbuf, 4 * DD, 4 * DD, 0);
        k_gemm_skinny<<<gSkinny4, 128, 0, stream>>>(wh_bf, wWhh_bf, gbuf, DD, 4 * DD, 1);
        k_lstm<<<gLstm, 256, 0, stream>>>(gbuf, word_bih, word_bhh, wh, wc, wh_bf, y_train, t, 1);
        // rule LSTM
        k_concat_rule<<<gElem, 256, 0, stream>>>(yemb_t, rfeed, wh_bf, xr_bf);
        k_gemm_skinny<<<gSkinny4, 128, 0, stream>>>(xr_bf, rWih_bf, gbuf, 4 * DD, 4 * DD, 0);
        k_gemm_skinny<<<gSkinny4, 128, 0, stream>>>(rh_bf, rWhh_bf, gbuf, DD, 4 * DD, 1);
        k_lstm<<<gLstm, 256, 0, stream>>>(gbuf, rule_bih, rule_bhh, rh, rc, rh_bf, y_train, t, 0);
        // attention contexts (become next step's feeds)
        k_attention<<<BB, 256, 0, stream>>>(rh, x_enc_k, x_enc, x_mask, rfeed);
        k_attention<<<BB, 256, 0, stream>>>(wh, x_enc_k, x_enc, x_mask, wfeed);
        // readout pre-activations
        k_concat_inp<<<gInp, 256, 0, stream>>>(rh_bf, rfeed, wh_bf, wfeed, inp_bf);
        k_gemm_skinny<<<gSkinny1, 128, 0, stream>>>(inp_bf, rro_bf, gro, 6 * DD, DD, 0);
        k_tanh_bf<<<gLstm, 256, 0, stream>>>(gro, rpre_bf + (size_t)t * BB * DD, BB * DD);
        k_gemm_skinny<<<gSkinny1, 128, 0, stream>>>(inp_bf, wro_bf, gro, 6 * DD, DD, 0);
        k_tanh_bf<<<gLstm, 256, 0, stream>>>(gro, wpre_bf + (size_t)t * BB * DD, BB * DD);
    }

    // logits: word region [0,WV) from wpre, masked when sm==0;
    //         rule region [WV,V) from rpre (readout rows WV..V), masked when sm==1.
    dim3 gw(WV / 64, (TT * BB) / 32);
    k_gemm_logits<<<gw, 256, 0, stream>>>(wpre_bf, rout_bf, out, score_mask, WV, 0, 1);
    dim3 gr(((VV - WV) + 63) / 64, (TT * BB) / 32);
    k_gemm_logits<<<gr, 256, 0, stream>>>(rpre_bf, rout_bf + (size_t)WV * DD, out, score_mask,
                                          VV - WV, WV, 0);
}
